// Model_3796751090164
// MI455X (gfx1250) — matrix operand order repaired and checked by reading, not compiled, not run on hardware
//
#include <hip/hip_runtime.h>
#include <hip/hip_bf16.h>

// ---------------------------------------------------------------------------
// CDNA5 / gfx1250: wave32, WMMA 16x16x32 bf16 (fp32 accumulate),
// async global->LDS staging (ASYNCcnt), global_prefetch for edge gathers.
// ---------------------------------------------------------------------------
typedef __attribute__((ext_vector_type(16))) __bf16 v16bf;
typedef __attribute__((ext_vector_type(8)))  float  v8f;

#define DEV __device__ __forceinline__

static DEV __bf16 tobf(float f) { return (__bf16)f; }

static DEV v8f wmma_bf16(v16bf a, v16bf b, v8f c) {
  return __builtin_amdgcn_wmma_f32_16x16x32_bf16(false, a, false, b,
                                                 (short)0, c, false, false);
}

// Async copy 16 bytes global -> LDS (GLOBAL_LOAD_ASYNC_TO_LDS_B128, ASYNCcnt).
// Per-lane LDS byte offset (low 32 bits of the generic LDS pointer) and
// per-lane 64-bit global address.
static DEV void async_copy_b128(unsigned lds_byte_off, const float* gsrc) {
  asm volatile("global_load_async_to_lds_b128 %0, %1, off"
               :
               : "v"(lds_byte_off), "v"(gsrc)
               : "memory");
}
static DEV void wait_async_all() {
  asm volatile("s_wait_asynccnt 0x0" ::: "memory");
}

// A-fragment (16x32 bf16, row-major source with row stride = K):
// lane holds two contiguous 8-element K-runs: [h*8, h*8+8) and [16+h*8, ...).
static DEV v16bf load_frag_a(const __bf16* p /* row base + kk */, int half) {
  union { v16bf v; uint4 q[2]; } u;
  u.q[0] = *(const uint4*)(p + half * 8);
  u.q[1] = *(const uint4*)(p + 16 + half * 8);
  return u.v;
}
// B-fragment (32x16 bf16) from a K-major row: two 8-element K-runs: [h*8, h*8+8) and [16+h*8, ...).
static DEV v16bf load_frag_b(const __bf16* p /* col base + kk */, int half) {
  union { v16bf v; uint4 q[2]; } u;
  u.q[0] = *(const uint4*)(p + half * 8);
  u.q[1] = *(const uint4*)(p + 16 + half * 8);
  return u.v;
}

// ---------------------------------------------------------------------------
// Weight prep: W [K][N] fp32  ->  Wt[n*ldt + kofs + k] bf16  (K-major).
// ---------------------------------------------------------------------------
__global__ void transpose_bf16_kernel(const float* __restrict__ W,
                                      __bf16* __restrict__ Wt,
                                      int K, int N, int ldt, int kofs) {
  int idx = blockIdx.x * blockDim.x + threadIdx.x;
  if (idx >= K * N) return;
  int k = idx / N, n = idx % N;
  Wt[(size_t)n * ldt + kofs + k] = tobf(W[(size_t)k * N + n]);
}

// ---------------------------------------------------------------------------
// Edge mean-aggregation: grid-stride, one wave per edge; float4 row loads;
// hardware f32 atomics into L2-resident accumulators; prefetch next source
// row (global_prefetch_b8) to hide random-gather latency.
// ---------------------------------------------------------------------------
template <int F>
__global__ void edge_agg_kernel(const float* __restrict__ x,
                                const int* __restrict__ src,
                                const int* __restrict__ dst,
                                float* __restrict__ agg,
                                float* __restrict__ deg, int E) {
  const int lane   = threadIdx.x & 31;
  const int wave0  = (blockIdx.x * blockDim.x + threadIdx.x) >> 5;
  const int nwaves = (gridDim.x * blockDim.x) >> 5;
  for (int e = wave0; e < E; e += nwaves) {
    int s = src[e];
    int d = dst[e];
    int en = e + nwaves;
    if (en < E) {
      int sn = src[en];  // uniform -> scalar load
      __builtin_prefetch(x + (size_t)sn * F + lane * (F / 32), 0, 0);
    }
    if (lane == 0) unsafeAtomicAdd(&deg[d], 1.0f);
    const float* xs = x + (size_t)s * F;
    float* ad = agg + (size_t)d * F;
#pragma unroll
    for (int c0 = 0; c0 < F; c0 += 128) {
      float4 v = *(const float4*)(xs + c0 + lane * 4);
      unsafeAtomicAdd(ad + c0 + lane * 4 + 0, v.x);
      unsafeAtomicAdd(ad + c0 + lane * 4 + 1, v.y);
      unsafeAtomicAdd(ad + c0 + lane * 4 + 2, v.z);
      unsafeAtomicAdd(ad + c0 + lane * 4 + 3, v.w);
    }
  }
}

// ---------------------------------------------------------------------------
// SAGE layer: out = LayerNorm(relu(mean @ Wl + bl + x @ Wr)) for 16 nodes.
// Raw fp32 [agg | x] rows staged to LDS with async B128 copies (no VGPR
// round-trip), then deg-scaled + converted to bf16 in LDS. B from K-major
// bf16 weights (B128 loads, L2 hits). 8 waves own disjoint column tiles.
// ---------------------------------------------------------------------------
template <int F_IN, int F_OUT>
__global__ void sage_layer_kernel(const float* __restrict__ x,
                                  const float* __restrict__ agg,
                                  const float* __restrict__ deg,
                                  const __bf16* __restrict__ Wt,  // [F_OUT][2*F_IN]
                                  const float* __restrict__ bl,
                                  const float* __restrict__ ln_g,
                                  const float* __restrict__ ln_b,
                                  float* __restrict__ out) {
  constexpr int K = 2 * F_IN;
  __shared__ float  sRaw[16 * K];
  __shared__ __bf16 sA[16 * K];
  __shared__ float  sO[16 * F_OUT];

  const int tid   = threadIdx.x;
  const int lane  = tid & 31;
  const int wid   = tid >> 5;
  const int node0 = blockIdx.x * 16;
  const int half  = lane >> 4;
  const int mrow  = lane & 15;
  const int ncol  = lane & 15;

  // Async-stage raw fp32 rows: sRaw[r][k] = (k<F_IN) ? agg[node][k] : x[node][k-F_IN]
  constexpr int CHUNKS = 16 * K / 4;  // multiple of 256 -> no divergence
  for (int c = tid; c < CHUNKS; c += 256) {
    int i = c * 4;
    int r = i / K, k = i % K;
    int node = node0 + r;
    const float* g = (k < F_IN) ? (agg + (size_t)node * F_IN + k)
                                : (x + (size_t)node * F_IN + (k - F_IN));
    async_copy_b128((unsigned)(uintptr_t)(sRaw + i), g);
  }
  wait_async_all();
  __syncthreads();

  // Scale mean part by 1/deg and convert to bf16 (LDS -> LDS, on-chip).
  for (int i = tid * 4; i < 16 * K; i += 256 * 4) {
    int r = i / K, k = i % K;  // k, F_IN multiples of 4 -> no region straddle
    float4 v = *(const float4*)(sRaw + i);
    if (k < F_IN) {
      float inv = 1.0f / fmaxf(deg[node0 + r], 1.0f);
      v.x *= inv; v.y *= inv; v.z *= inv; v.w *= inv;
    }
    sA[i + 0] = tobf(v.x); sA[i + 1] = tobf(v.y);
    sA[i + 2] = tobf(v.z); sA[i + 3] = tobf(v.w);
  }
  __syncthreads();

  for (int nt = wid; nt < F_OUT / 16; nt += 8) {
    int col = nt * 16 + ncol;
    const __bf16* wcol = Wt + (size_t)col * K;
    v8f acc = {};
    for (int kk = 0; kk < K; kk += 32) {
      v16bf afrag = load_frag_a(sA + mrow * K + kk, half);
      v16bf bfrag = load_frag_b(wcol + kk, half);
      acc = wmma_bf16(afrag, bfrag, acc);
    }
    float bias = bl[col];
#pragma unroll
    for (int r = 0; r < 8; ++r) {
      // D layout: VGPR r holds row M = r + 8*(lane/16).
      sO[(half * 8 + r) * F_OUT + col] = fmaxf(acc[r] + bias, 0.0f);
    }
  }
  __syncthreads();

  // LayerNorm: one wave per row (rows wid, wid+8), shuffle reduction.
  for (int r = wid; r < 16; r += 8) {
    float s1 = 0.f, s2 = 0.f;
    for (int c = lane; c < F_OUT; c += 32) {
      float v = sO[r * F_OUT + c];
      s1 += v; s2 += v * v;
    }
#pragma unroll
    for (int off = 16; off > 0; off >>= 1) {
      s1 += __shfl_down(s1, off, 32);
      s2 += __shfl_down(s2, off, 32);
    }
    s1 = __shfl(s1, 0, 32);
    s2 = __shfl(s2, 0, 32);
    float mu  = s1 / (float)F_OUT;
    float var = s2 / (float)F_OUT - mu * mu;
    float inv = rsqrtf(var + 1e-5f);
    size_t base = (size_t)(node0 + r) * F_OUT;
    for (int c = lane; c < F_OUT; c += 32)
      out[base + c] = (sO[r * F_OUT + c] - mu) * inv * ln_g[c] + ln_b[c];
  }
}

// ---------------------------------------------------------------------------
// MLP head: relu(relu(relu(X@W1+b1)@W2+b2)@W3+b3), 3 GEMMs fused, activations
// ping-ponged through LDS; weights pre-transposed bf16; output stored bf16.
// ---------------------------------------------------------------------------
__global__ void mlp_head_kernel(const float* __restrict__ X,
                                const __bf16* __restrict__ W1t,  // [256][128]
                                const float* __restrict__ b1,
                                const __bf16* __restrict__ W2t,  // [128][256]
                                const float* __restrict__ b2,
                                const __bf16* __restrict__ W3t,  // [64][128]
                                const float* __restrict__ b3,
                                __bf16* __restrict__ outk) {
  __shared__ __bf16 sIn[16 * 256];
  __shared__ float  sAcc[16 * 256];
  const int tid   = threadIdx.x;
  const int lane  = tid & 31;
  const int wid   = tid >> 5;
  const int node0 = blockIdx.x * 16;
  const int half  = lane >> 4;
  const int mrow  = lane & 15;
  const int ncol  = lane & 15;

  for (int i = tid * 4; i < 16 * 128; i += blockDim.x * 4) {
    float4 v = *(const float4*)(X + (size_t)node0 * 128 + i);
    sIn[i + 0] = tobf(v.x); sIn[i + 1] = tobf(v.y);
    sIn[i + 2] = tobf(v.z); sIn[i + 3] = tobf(v.w);
  }
  __syncthreads();

  // Layer 1: [16,128] @ [128,256]
  for (int nt = wid; nt < 16; nt += 8) {
    int col = nt * 16 + ncol;
    const __bf16* wcol = W1t + (size_t)col * 128;
    v8f acc = {};
    for (int kk = 0; kk < 128; kk += 32)
      acc = wmma_bf16(load_frag_a(sIn + mrow * 128 + kk, half),
                      load_frag_b(wcol + kk, half), acc);
    float bias = b1[col];
#pragma unroll
    for (int r = 0; r < 8; ++r)
      sAcc[(half * 8 + r) * 256 + col] = fmaxf(acc[r] + bias, 0.f);
  }
  __syncthreads();
  for (int i = tid; i < 16 * 256; i += blockDim.x) sIn[i] = tobf(sAcc[i]);
  __syncthreads();

  // Layer 2: [16,256] @ [256,128] (8 waves, 8 tiles)
  {
    int col = wid * 16 + ncol;
    const __bf16* wcol = W2t + (size_t)col * 256;
    v8f acc = {};
    for (int kk = 0; kk < 256; kk += 32)
      acc = wmma_bf16(load_frag_a(sIn + mrow * 256 + kk, half),
                      load_frag_b(wcol + kk, half), acc);
    float bias = b2[col];
#pragma unroll
    for (int r = 0; r < 8; ++r)
      sAcc[(half * 8 + r) * 128 + col] = fmaxf(acc[r] + bias, 0.f);
  }
  __syncthreads();
  for (int i = tid; i < 16 * 128; i += blockDim.x) sIn[i] = tobf(sAcc[i]);
  __syncthreads();

  // Layer 3: [16,128] @ [128,64] -> bf16 global
  if (wid < 4) {
    int col = wid * 16 + ncol;
    const __bf16* wcol = W3t + (size_t)col * 128;
    v8f acc = {};
    for (int kk = 0; kk < 128; kk += 32)
      acc = wmma_bf16(load_frag_a(sIn + mrow * 128 + kk, half),
                      load_frag_b(wcol + kk, half), acc);
    float bias = b3[col];
#pragma unroll
    for (int r = 0; r < 8; ++r)
      outk[(size_t)(node0 + half * 8 + r) * 64 + col] =
          tobf(fmaxf(acc[r] + bias, 0.f));
  }
}

// ---------------------------------------------------------------------------
// Final GEMM: out[M,D] = Xk @ Yk^T, K=64 bf16 -> fp32. Store-bound (512MB);
// 8 waves/block, each wave one 16x64 tile; all operand loads are B128.
// ---------------------------------------------------------------------------
__global__ void final_gemm_kernel(const __bf16* __restrict__ Xk,
                                  const __bf16* __restrict__ Yk,
                                  float* __restrict__ out, int Dtot) {
  const int lane = threadIdx.x & 31;
  const int wid  = threadIdx.x >> 5;
  const int half = lane >> 4;
  const int mrow = lane & 15;
  const int ncol = lane & 15;
  const int row0 = blockIdx.y * 128 + wid * 16;
  const int col0 = blockIdx.x * 64;

  v8f acc[4] = {};
#pragma unroll
  for (int kk = 0; kk < 64; kk += 32) {
    v16bf a = load_frag_a(Xk + (size_t)(row0 + mrow) * 64 + kk, half);
#pragma unroll
    for (int t = 0; t < 4; ++t) {
      v16bf b = load_frag_b(Yk + (size_t)(col0 + t * 16 + ncol) * 64 + kk, half);
      acc[t] = wmma_bf16(a, b, acc[t]);
    }
  }
#pragma unroll
  for (int t = 0; t < 4; ++t) {
    int col = col0 + t * 16 + ncol;
#pragma unroll
    for (int r = 0; r < 8; ++r)
      out[(size_t)(row0 + half * 8 + r) * Dtot + col] = acc[t][r];
  }
}

// ---------------------------------------------------------------------------
extern "C" void kernel_launch(void* const* d_in, const int* in_sizes, int n_in,
                              void* d_out, int out_size, void* d_ws, size_t ws_size,
                              hipStream_t stream) {
  constexpr int M = 16384, Dn = 8192;
  constexpr int E_MM = 524288, E_DD = 262144;

  const float* x_m = (const float*)d_in[0];
  const float* x_d = (const float*)d_in[1];
  const int*   mm  = (const int*)d_in[2];
  const int*   dd  = (const int*)d_in[3];
  const float *sx1_Wl=(const float*)d_in[4],  *sx1_bl=(const float*)d_in[5],  *sx1_Wr=(const float*)d_in[6];
  const float *sx2_Wl=(const float*)d_in[7],  *sx2_bl=(const float*)d_in[8],  *sx2_Wr=(const float*)d_in[9];
  const float *sy1_Wl=(const float*)d_in[10], *sy1_bl=(const float*)d_in[11], *sy1_Wr=(const float*)d_in[12];
  const float *sy2_Wl=(const float*)d_in[13], *sy2_bl=(const float*)d_in[14], *sy2_Wr=(const float*)d_in[15];
  const float *nx1_g=(const float*)d_in[16], *nx1_b=(const float*)d_in[17];
  const float *nx2_g=(const float*)d_in[18], *nx2_b=(const float*)d_in[19];
  const float *ny1_g=(const float*)d_in[20], *ny1_b=(const float*)d_in[21];
  const float *ny2_g=(const float*)d_in[22], *ny2_b=(const float*)d_in[23];
  const float *lx1_W=(const float*)d_in[24], *lx1_b=(const float*)d_in[25];
  const float *lx2_W=(const float*)d_in[26], *lx2_b=(const float*)d_in[27];
  const float *lx3_W=(const float*)d_in[28], *lx3_b=(const float*)d_in[29];
  const float *ly1_W=(const float*)d_in[30], *ly1_b=(const float*)d_in[31];
  const float *ly2_W=(const float*)d_in[32], *ly2_b=(const float*)d_in[33];
  const float *ly3_W=(const float*)d_in[34], *ly3_b=(const float*)d_in[35];

  // Workspace carve-out (~64 MB total).
  char* w = (char*)d_ws;
  auto take = [&](size_t bytes) {
    char* p = w;
    w += (bytes + 255) & ~size_t(255);
    return (void*)p;
  };
  float*  agg_x = (float*)take((size_t)M  * 256 * 4);
  float*  deg_m = (float*)take((size_t)M  * 4);
  float*  agg_y = (float*)take((size_t)Dn * 256 * 4);
  float*  deg_d = (float*)take((size_t)Dn * 4);
  float*  X1    = (float*)take((size_t)M  * 256 * 4);
  float*  Xf    = (float*)take((size_t)M  * 128 * 4);
  float*  Y1    = (float*)take((size_t)Dn * 256 * 4);
  float*  Yf    = (float*)take((size_t)Dn * 128 * 4);
  __bf16* Xk    = (__bf16*)take((size_t)M  * 64 * 2);
  __bf16* Yk    = (__bf16*)take((size_t)Dn * 64 * 2);
  // K-major bf16 weights.
  __bf16* WX1t  = (__bf16*)take((size_t)256 * 256 * 2);  // [256][256]
  __bf16* WX2t  = (__bf16*)take((size_t)128 * 512 * 2);  // [128][512]
  __bf16* WY1t  = (__bf16*)take((size_t)256 * 256 * 2);
  __bf16* WY2t  = (__bf16*)take((size_t)128 * 512 * 2);
  __bf16* HX1t  = (__bf16*)take((size_t)256 * 128 * 2);
  __bf16* HX2t  = (__bf16*)take((size_t)128 * 256 * 2);
  __bf16* HX3t  = (__bf16*)take((size_t)64  * 128 * 2);
  __bf16* HY1t  = (__bf16*)take((size_t)256 * 128 * 2);
  __bf16* HY2t  = (__bf16*)take((size_t)128 * 256 * 2);
  __bf16* HY3t  = (__bf16*)take((size_t)64  * 128 * 2);

  const int* mm_src = mm;  const int* mm_dst = mm + E_MM;
  const int* dd_src = dd;  const int* dd_dst = dd + E_DD;

  dim3 blk(256);
  auto tgrid = [](int K, int N) { return dim3((K * N + 255) / 256); };

  // ----- weight prep (transpose + bf16 cast) -----
  transpose_bf16_kernel<<<tgrid(128, 256), blk, 0, stream>>>(sx1_Wl, WX1t, 128, 256, 256, 0);
  transpose_bf16_kernel<<<tgrid(128, 256), blk, 0, stream>>>(sx1_Wr, WX1t, 128, 256, 256, 128);
  transpose_bf16_kernel<<<tgrid(256, 128), blk, 0, stream>>>(sx2_Wl, WX2t, 256, 128, 512, 0);
  transpose_bf16_kernel<<<tgrid(256, 128), blk, 0, stream>>>(sx2_Wr, WX2t, 256, 128, 512, 256);
  transpose_bf16_kernel<<<tgrid(128, 256), blk, 0, stream>>>(sy1_Wl, WY1t, 128, 256, 256, 0);
  transpose_bf16_kernel<<<tgrid(128, 256), blk, 0, stream>>>(sy1_Wr, WY1t, 128, 256, 256, 128);
  transpose_bf16_kernel<<<tgrid(256, 128), blk, 0, stream>>>(sy2_Wl, WY2t, 256, 128, 512, 0);
  transpose_bf16_kernel<<<tgrid(256, 128), blk, 0, stream>>>(sy2_Wr, WY2t, 256, 128, 512, 256);
  transpose_bf16_kernel<<<tgrid(128, 256), blk, 0, stream>>>(lx1_W, HX1t, 128, 256, 128, 0);
  transpose_bf16_kernel<<<tgrid(256, 128), blk, 0, stream>>>(lx2_W, HX2t, 256, 128, 256, 0);
  transpose_bf16_kernel<<<tgrid(128, 64),  blk, 0, stream>>>(lx3_W, HX3t, 128, 64, 128, 0);
  transpose_bf16_kernel<<<tgrid(128, 256), blk, 0, stream>>>(ly1_W, HY1t, 128, 256, 128, 0);
  transpose_bf16_kernel<<<tgrid(256, 128), blk, 0, stream>>>(ly2_W, HY2t, 256, 128, 256, 0);
  transpose_bf16_kernel<<<tgrid(128, 64),  blk, 0, stream>>>(ly3_W, HY3t, 128, 64, 128, 0);

  const int EBLK = 8192;  // 65536 waves, grid-stride over edges

  // ----- x tower -----
  hipMemsetAsync(agg_x, 0, (size_t)M * 128 * 4, stream);
  hipMemsetAsync(deg_m, 0, (size_t)M * 4, stream);
  edge_agg_kernel<128><<<EBLK, blk, 0, stream>>>(x_m, mm_src, mm_dst, agg_x, deg_m, E_MM);
  sage_layer_kernel<128, 256><<<M / 16, blk, 0, stream>>>(
      x_m, agg_x, deg_m, WX1t, sx1_bl, nx1_g, nx1_b, X1);
  hipMemsetAsync(agg_x, 0, (size_t)M * 256 * 4, stream);
  hipMemsetAsync(deg_m, 0, (size_t)M * 4, stream);
  edge_agg_kernel<256><<<EBLK, blk, 0, stream>>>(X1, mm_src, mm_dst, agg_x, deg_m, E_MM);
  sage_layer_kernel<256, 128><<<M / 16, blk, 0, stream>>>(
      X1, agg_x, deg_m, WX2t, sx2_bl, nx2_g, nx2_b, Xf);

  // ----- y tower -----
  hipMemsetAsync(agg_y, 0, (size_t)Dn * 128 * 4, stream);
  hipMemsetAsync(deg_d, 0, (size_t)Dn * 4, stream);
  edge_agg_kernel<128><<<EBLK, blk, 0, stream>>>(x_d, dd_src, dd_dst, agg_y, deg_d, E_DD);
  sage_layer_kernel<128, 256><<<Dn / 16, blk, 0, stream>>>(
      x_d, agg_y, deg_d, WY1t, sy1_bl, ny1_g, ny1_b, Y1);
  hipMemsetAsync(agg_y, 0, (size_t)Dn * 256 * 4, stream);
  hipMemsetAsync(deg_d, 0, (size_t)Dn * 4, stream);
  edge_agg_kernel<256><<<EBLK, blk, 0, stream>>>(Y1, dd_src, dd_dst, agg_y, deg_d, E_DD);
  sage_layer_kernel<256, 128><<<Dn / 16, blk, 0, stream>>>(
      Y1, agg_y, deg_d, WY2t, sy2_bl, ny2_g, ny2_b, Yf);

  // ----- MLP heads -----
  mlp_head_kernel<<<M / 16, blk, 0, stream>>>(Xf, HX1t, lx1_b, HX2t, lx2_b, HX3t, lx3_b, Xk);
  mlp_head_kernel<<<Dn / 16, blk, 0, stream>>>(Yf, HY1t, ly1_b, HY2t, ly2_b, HY3t, ly3_b, Yk);

  // ----- final GEMM -----
  dim3 grid(Dn / 64, M / 128);
  final_gemm_kernel<<<grid, blk, 0, stream>>>(Xk, Yk, (float*)d_out, Dn);
}
